// RGCAT_68384469286919
// MI455X (gfx1250) — matrix operand order repaired and checked by reading, not compiled, not run on hardware
//
#include <hip/hip_runtime.h>

#define NN 2048
#define IN_F 512
#define HID 512
#define NH 8
#define DQK 64
#define DV 32
#define SPAD 8
#define SSTR (NN + SPAD)   // 2056 floats per LDS score row

typedef __bf16 bf16_t;
typedef __attribute__((ext_vector_type(8)))  __bf16 v8bf;
typedef __attribute__((ext_vector_type(16))) __bf16 v16bf;
typedef __attribute__((ext_vector_type(8)))  float  v8f;

static __device__ __forceinline__ bf16_t f32_to_bf16(float f) {
  unsigned u = __builtin_bit_cast(unsigned, f);
  unsigned r = (u + 0x7FFFu + ((u >> 16) & 1u)) >> 16;
  return __builtin_bit_cast(bf16_t, (unsigned short)r);
}

static __device__ __forceinline__ v8f wmma_bf16(v16bf a, v16bf b, v8f c) {
  return __builtin_amdgcn_wmma_f32_16x16x32_bf16(false, a, false, b,
                                                 (short)0, c, false, false);
}

// A-fragment: 16x32 bf16, A row-major (lda elems). Per ISA layout:
// lane<16: elems0-7 K=k0..k0+7, elems8-15 K=k0+16..k0+23 ; lane>=16: +8
static __device__ __forceinline__ v16bf load_a_frag(const bf16_t* A, int lda,
                                                    int m0, int k0, int lane) {
  int m  = m0 + (lane & 15);
  int kh = (lane >> 4) << 3;                 // 0 or 8
  const bf16_t* p = A + (size_t)m * lda + k0 + kh;
  v8bf lo = *(const v8bf*)p;
  v8bf hi = *(const v8bf*)(p + 16);
  return __builtin_shufflevector(lo, hi, 0,1,2,3,4,5,6,7,8,9,10,11,12,13,14,15);
}

// B-fragment: 32x16 bf16 where B[k][n] = Bt[n][k], Bt row-major (ldb elems).
// lane<16: elems0-7 K=k0..k0+7, elems8-15 K=k0+16..k0+23 ; lane>=16: +8 ; N = n0 + lane&15
static __device__ __forceinline__ v16bf load_bt_frag(const bf16_t* Bt, int ldb,
                                                     int n0, int k0, int lane) {
  int n  = n0 + (lane & 15);
  int kh = (lane >> 4) << 3;                 // 0 or 8
  const bf16_t* p = Bt + (size_t)n * ldb + k0 + kh;
  v8bf lo = *(const v8bf*)p;
  v8bf hi = *(const v8bf*)(p + 16);
  return __builtin_shufflevector(lo, hi, 0,1,2,3,4,5,6,7,8,9,10,11,12,13,14,15);
}

static __device__ __forceinline__ void store_tile(void* Cp, int ldc, int m0, int n0,
                                                  const float* bias, v8f acc,
                                                  int lane, int mode) {
  const int mBase = m0 + ((lane >> 4) << 3);
  const int n     = n0 + (lane & 15);
  const float bv  = bias ? bias[n] : 0.0f;
  if (mode == 0) {
    float* C = (float*)Cp;
#pragma unroll
    for (int v = 0; v < 8; ++v) C[(size_t)(mBase + v) * ldc + n] = acc[v] + bv;
  } else if (mode == 1) {
    bf16_t* C = (bf16_t*)Cp;
#pragma unroll
    for (int v = 0; v < 8; ++v) C[(size_t)(mBase + v) * ldc + n] = f32_to_bf16(acc[v] + bv);
  } else {
    bf16_t* C = (bf16_t*)Cp;
#pragma unroll
    for (int v = 0; v < 8; ++v) C[(size_t)n * ldc + (mBase + v)] = f32_to_bf16(acc[v] + bv);
  }
}

// ------------- generic WMMA GEMM: C = A * Bt^T + bias -------------
// 2x2 register macro-tile per wave (32x32 output), fragment reuse: 4 loads -> 4 WMMA.
// mode 0: f32 row-major   mode 1: bf16 row-major   mode 2: bf16 transposed
__global__ __launch_bounds__(256) void gemm_wmma(
    const bf16_t* __restrict__ A, int lda,
    const bf16_t* __restrict__ Bt, int ldb,
    const float* __restrict__ bias,
    void* __restrict__ Cp, int ldc,
    int M, int Ncol, int K, int mode) {
  const int lane = threadIdx.x & 31;
  const int wave = threadIdx.x >> 5;
  const int ntN  = Ncol >> 5;
  const int tile = blockIdx.x * 8 + wave;
  if (tile >= (M >> 5) * ntN) return;
  const int m0 = (tile / ntN) << 5;
  const int n0 = (tile % ntN) << 5;

  v8f c00 = {0.f,0.f,0.f,0.f,0.f,0.f,0.f,0.f};
  v8f c01 = c00, c10 = c00, c11 = c00;
  for (int k0 = 0; k0 < K; k0 += 32) {
    if (k0 + 64 < K) {
      __builtin_prefetch(A + (size_t)(m0 + (lane & 15)) * lda + k0 + 64, 0, 1);
      __builtin_prefetch(Bt + (size_t)(n0 + (lane & 15)) * ldb + k0 + 64, 0, 1);
    }
    v16bf a0 = load_a_frag(A, lda, m0,      k0, lane);
    v16bf a1 = load_a_frag(A, lda, m0 + 16, k0, lane);
    v16bf b0 = load_bt_frag(Bt, ldb, n0,      k0, lane);
    v16bf b1 = load_bt_frag(Bt, ldb, n0 + 16, k0, lane);
    c00 = wmma_bf16(a0, b0, c00);
    c01 = wmma_bf16(a0, b1, c01);
    c10 = wmma_bf16(a1, b0, c10);
    c11 = wmma_bf16(a1, b1, c11);
  }
  store_tile(Cp, ldc, m0,      n0,      bias, c00, lane, mode);
  store_tile(Cp, ldc, m0,      n0 + 16, bias, c01, lane, mode);
  store_tile(Cp, ldc, m0 + 16, n0,      bias, c10, lane, mode);
  store_tile(Cp, ldc, m0 + 16, n0 + 16, bias, c11, lane, mode);
}

// ------------- fused scores + cross-gather + masked softmax -------------
// block = (row block of 16, head). LDS: S1,S2 f32 [16][SSTR] = ~257 KB (<320 KB/WGP)
__global__ __launch_bounds__(256) void att_scores(
    const bf16_t* __restrict__ q1, const bf16_t* __restrict__ k1,
    const bf16_t* __restrict__ q2, const bf16_t* __restrict__ k2,
    const int* __restrict__ c2n, const int* __restrict__ n2c,
    const int* __restrict__ edge, const int* __restrict__ conj,
    bf16_t* __restrict__ a1, bf16_t* __restrict__ a2, float scale) {
  extern __shared__ float smem[];          // [2][16][SSTR]
  const int i0   = blockIdx.x << 4;
  const int h    = blockIdx.y;
  const int lane = threadIdx.x & 31;
  const int wave = threadIdx.x >> 5;
  const int colq = h * DQK;

  // Q fragments are loop-invariant: hoist (s in {1,2}) x (k0 in {0,32})
  const v16bf aq10 = load_a_frag(q1, HID, i0, colq,      lane);
  const v16bf aq11 = load_a_frag(q1, HID, i0, colq + 32, lane);
  const v16bf aq20 = load_a_frag(q2, HID, i0, colq,      lane);
  const v16bf aq21 = load_a_frag(q2, HID, i0, colq + 32, lane);

  // Phase 1: S1 = scale*Q1 K1^T, S2 = scale*Q2 K2^T  (16 x 2048 each)
  const int rb = (lane >> 4) << 3;
  for (int jt = wave; jt < (NN >> 4); jt += 8) {
    const int j0 = jt << 4;
    const int c  = j0 + (lane & 15);
    v8f acc = {0.f,0.f,0.f,0.f,0.f,0.f,0.f,0.f};
    acc = wmma_bf16(aq10, load_bt_frag(k1, HID, j0, colq,      lane), acc);
    acc = wmma_bf16(aq11, load_bt_frag(k1, HID, j0, colq + 32, lane), acc);
#pragma unroll
    for (int v = 0; v < 8; ++v) smem[(rb + v) * SSTR + c] = acc[v] * scale;
    v8f acc2 = {0.f,0.f,0.f,0.f,0.f,0.f,0.f,0.f};
    acc2 = wmma_bf16(aq20, load_bt_frag(k2, HID, j0, colq,      lane), acc2);
    acc2 = wmma_bf16(aq21, load_bt_frag(k2, HID, j0, colq + 32, lane), acc2);
#pragma unroll
    for (int v = 0; v < 8; ++v) smem[16 * SSTR + (rb + v) * SSTR + c] = acc2[v] * scale;
  }
  __syncthreads();

  // Phase 2: online max/sum per row (16 lanes per row)
  const int r = threadIdx.x >> 4;
  const int t = threadIdx.x & 15;
  const int i = i0 + r;
  const float* S1 = smem;
  const float* S2 = smem + 16 * SSTR;
  float m1 = -3.0e38f, s1 = 0.f, m2 = -3.0e38f, s2 = 0.f;
  for (int j = t; j < NN; j += 16) {
    const int g12 = c2n[(size_t)i * NN + j];
    const int g21 = n2c[(size_t)i * NN + j];
    const float mk1 = (float)(edge[(size_t)j * NN + i] - 1) * 10000.0f;
    const float mk2 = (float)(conj[(size_t)j * NN + i] - 1) * 10000.0f;
    const float p1 = S1[r * SSTR + j] + S2[r * SSTR + g12] + mk1;
    const float p2 = S2[r * SSTR + j] + S1[r * SSTR + g21] + mk2;
    float nm = fmaxf(m1, p1); s1 = s1 * __expf(m1 - nm) + __expf(p1 - nm); m1 = nm;
    nm = fmaxf(m2, p2);       s2 = s2 * __expf(m2 - nm) + __expf(p2 - nm); m2 = nm;
  }
#pragma unroll
  for (int off = 1; off < 16; off <<= 1) {
    float om = __shfl_xor(m1, off, 16), os = __shfl_xor(s1, off, 16);
    float nm = fmaxf(m1, om); s1 = s1 * __expf(m1 - nm) + os * __expf(om - nm); m1 = nm;
    om = __shfl_xor(m2, off, 16); os = __shfl_xor(s2, off, 16);
    nm = fmaxf(m2, om); s2 = s2 * __expf(m2 - nm) + os * __expf(om - nm); m2 = nm;
  }
  const float inv1 = 1.0f / s1, inv2 = 1.0f / s2;

  // Phase 3: recompute logits, emit normalized bf16 probabilities
  bf16_t* A1 = a1 + ((size_t)h * NN + i) * NN;
  bf16_t* A2 = a2 + ((size_t)h * NN + i) * NN;
  for (int j = t; j < NN; j += 16) {
    const int g12 = c2n[(size_t)i * NN + j];
    const int g21 = n2c[(size_t)i * NN + j];
    const float mk1 = (float)(edge[(size_t)j * NN + i] - 1) * 10000.0f;
    const float mk2 = (float)(conj[(size_t)j * NN + i] - 1) * 10000.0f;
    const float p1 = S1[r * SSTR + j] + S2[r * SSTR + g12] + mk1;
    const float p2 = S2[r * SSTR + j] + S1[r * SSTR + g21] + mk2;
    A1[j] = f32_to_bf16(__expf(p1 - m1) * inv1);
    A2[j] = f32_to_bf16(__expf(p2 - m2) * inv2);
  }
}

// ---------------- small elementwise kernels ----------------
__global__ void cvt_bf16(const float* __restrict__ in, bf16_t* __restrict__ out, int n) {
  int i = blockIdx.x * blockDim.x + threadIdx.x;
  if (i < n) out[i] = f32_to_bf16(in[i]);
}
__global__ void cvt_t_bf16(const float* __restrict__ in, bf16_t* __restrict__ out,
                           int rows, int cols) {
  int i = blockIdx.x * blockDim.x + threadIdx.x;
  if (i < rows * cols) {
    int rr = i / cols, cc = i - rr * cols;
    out[(size_t)cc * rows + rr] = f32_to_bf16(in[i]);   // out = in^T
  }
}
__global__ void update_x(const float* __restrict__ z, const float* __restrict__ xold,
                         const int* __restrict__ mask, float* __restrict__ xnew,
                         bf16_t* __restrict__ xnewb) {
  int idx = blockIdx.x * blockDim.x + threadIdx.x;
  if (idx < NN * IN_F) {
    float fm = (float)mask[idx >> 9];
    float v  = z[idx] * fm + xold[idx] * (1.0f - fm);
    xnew[idx]  = v;
    xnewb[idx] = f32_to_bf16(v);
  }
}
__global__ void classify_k(const float* __restrict__ x, const float* __restrict__ W,
                           const float* __restrict__ b, float* __restrict__ out) {
  int idx = blockIdx.x * blockDim.x + threadIdx.x;   // N*2
  if (idx >= NN * 2) return;
  int i = idx >> 1, c = idx & 1;
  float acc = b[c];
  const float* xr = x + (size_t)i * IN_F;
  for (int k = 0; k < IN_F; ++k) acc += xr[k] * W[k * 2 + c];
  out[idx] = acc;
}

// ---------------- host driver ----------------
extern "C" void kernel_launch(void* const* d_in, const int* in_sizes, int n_in,
                              void* d_out, int out_size, void* d_ws, size_t ws_size,
                              hipStream_t stream) {
  (void)in_sizes; (void)n_in; (void)out_size; (void)ws_size;
  const float* x_in = (const float*)d_in[0];
  const int* edge   = (const int*)d_in[1];
  const int* conj   = (const int*)d_in[2];
  const int* n2c    = (const int*)d_in[3];
  const int* c2n    = (const int*)d_in[4];
  const int* maskp  = (const int*)d_in[5];
  auto P = [&](int i) { return (const float*)d_in[i]; };

  char* w = (char*)d_ws;
  size_t off = 0;
  auto alloc = [&](size_t bytes) -> void* {
    void* p = w + off;
    off = (off + bytes + 255) & ~(size_t)255;
    return p;
  };
  bf16_t* xb   = (bf16_t*)alloc((size_t)NN * IN_F * 2);
  bf16_t* wq1t = (bf16_t*)alloc(512 * 512 * 2);
  bf16_t* wk1t = (bf16_t*)alloc(512 * 512 * 2);
  bf16_t* wq2t = (bf16_t*)alloc(512 * 512 * 2);
  bf16_t* wk2t = (bf16_t*)alloc(512 * 512 * 2);
  bf16_t* wv1t = (bf16_t*)alloc(512 * 256 * 2);
  bf16_t* wv2t = (bf16_t*)alloc(512 * 256 * 2);
  bf16_t* wfct = (bf16_t*)alloc(512 * 512 * 2);
  bf16_t* q1b  = (bf16_t*)alloc((size_t)NN * HID * 2);
  bf16_t* k1b  = (bf16_t*)alloc((size_t)NN * HID * 2);
  bf16_t* q2b  = (bf16_t*)alloc((size_t)NN * HID * 2);
  bf16_t* k2b  = (bf16_t*)alloc((size_t)NN * HID * 2);
  bf16_t* v1t  = (bf16_t*)alloc((size_t)256 * NN * 2);  // transposed (256 x N)
  bf16_t* v2t  = (bf16_t*)alloc((size_t)256 * NN * 2);
  bf16_t* a1   = (bf16_t*)alloc((size_t)NH * NN * NN * 2);
  bf16_t* a2   = (bf16_t*)alloc((size_t)NH * NN * NN * 2);
  bf16_t* obf  = (bf16_t*)alloc((size_t)NN * HID * 2);
  float*  z    = (float*)alloc((size_t)NN * IN_F * 4);
  float*  x1   = (float*)alloc((size_t)NN * IN_F * 4);
  float*  x2   = (float*)alloc((size_t)NN * IN_F * 4);
  float*  outf = (float*)d_out;

  auto gemm = [&](const bf16_t* A, int lda, const bf16_t* Bt, int ldb,
                  const float* bias, void* C, int ldc, int M, int Nc, int K, int mode) {
    int tiles = (M / 32) * (Nc / 32);
    gemm_wmma<<<(tiles + 7) / 8, 256, 0, stream>>>(A, lda, Bt, ldb, bias, C, ldc, M, Nc, K, mode);
  };
  auto cT = [&](const float* in, bf16_t* o, int r, int c) {
    cvt_t_bf16<<<(r * c + 255) / 256, 256, 0, stream>>>(in, o, r, c);
  };

  cvt_bf16<<<(NN * IN_F + 255) / 256, 256, 0, stream>>>(x_in, xb, NN * IN_F);
  classify_k<<<(NN * 2 + 255) / 256, 256, 0, stream>>>(x_in, P(34), P(35), outf);

  for (int l = 0; l < 2; ++l) {
    const int base = l ? 18 : 6;
    const float* fcW = P(l ? 30 : 32);
    const float* fcB = P(l ? 31 : 33);
    cT(P(base + 0), wq1t, 512, 512);
    cT(P(base + 2), wk1t, 512, 512);
    cT(P(base + 4), wv1t, 512, 256);
    cT(P(base + 6), wq2t, 512, 512);
    cT(P(base + 8), wk2t, 512, 512);
    cT(P(base + 10), wv2t, 512, 256);
    cT(fcW, wfct, 512, 512);

    gemm(xb, 512, wq1t, 512, P(base + 1),  q1b, 512,  NN, 512, 512, 1);
    gemm(xb, 512, wk1t, 512, P(base + 3),  k1b, 512,  NN, 512, 512, 1);
    gemm(xb, 512, wv1t, 512, P(base + 5),  v1t, NN,   NN, 256, 512, 2);
    gemm(xb, 512, wq2t, 512, P(base + 7),  q2b, 512,  NN, 512, 512, 1);
    gemm(xb, 512, wk2t, 512, P(base + 9),  k2b, 512,  NN, 512, 512, 1);
    gemm(xb, 512, wv2t, 512, P(base + 11), v2t, NN,   NN, 256, 512, 2);

    size_t smem = (size_t)2 * 16 * SSTR * sizeof(float);
    att_scores<<<dim3(NN / 16, NH), 256, smem, stream>>>(
        q1b, k1b, q2b, k2b, c2n, n2c,
        edge + (size_t)l * NN * NN, conj + (size_t)l * NN * NN, a1, a2, 0.125f);

    for (int h = 0; h < NH; ++h) {
      gemm(a1 + (size_t)h * NN * NN, NN, v1t + (size_t)h * DV * NN, NN,
           nullptr, obf + h * DV, HID, NN, DV, NN, 1);
      gemm(a2 + (size_t)h * NN * NN, NN, v2t + (size_t)h * DV * NN, NN,
           nullptr, obf + 256 + h * DV, HID, NN, DV, NN, 1);
    }

    gemm(obf, 512, wfct, 512, fcB, z, 512, NN, 512, 512, 0);

    const float* xold = l ? (const float*)x1 : x_in;
    float* xnew = l ? x2 : x1;
    update_x<<<(NN * IN_F + 255) / 256, 256, 0, stream>>>(z, xold, maskp + l * NN, xnew, xb);
    classify_k<<<(NN * 2 + 255) / 256, 256, 0, stream>>>(xnew, P(34), P(35),
                                                         outf + (size_t)(l + 1) * NN * 2);
  }
}
